// VEConv_16449724744297
// MI455X (gfx1250) — matrix operand order repaired and checked by reading, not compiled, not run on hardware
//
#include <hip/hip_runtime.h>

typedef __attribute__((ext_vector_type(16))) _Float16 v16h;
typedef __attribute__((ext_vector_type(8)))  float    v8f;

#define RBF_DIM 128
#define DIM     64

// Load 16 contiguous f32 (this lane's half of one 32-K A chunk) and pack to the
// v16h A-fragment in contiguous K order (matching the B reads):
//   v16h elems 0..7  -> VGPR0..3 : K = 32c + 16*half + {0..7}
//   v16h elems 8..15 -> VGPR4..7 : K = 32c + 16*half + 8 + {0..7}
__device__ __forceinline__ v16h load_A_f32(const float* __restrict__ rowp, int c, int half) {
    const float* p0 = rowp + 32 * c + 16 * half;
    const float* p1 = p0 + 8;
    float4 x0 = *reinterpret_cast<const float4*>(p0);
    float4 x1 = *reinterpret_cast<const float4*>(p0 + 4);
    float4 y0 = *reinterpret_cast<const float4*>(p1);
    float4 y1 = *reinterpret_cast<const float4*>(p1 + 4);
    v16h a;
    a[0]  = (_Float16)x0.x; a[1]  = (_Float16)x0.y; a[2]  = (_Float16)x0.z; a[3]  = (_Float16)x0.w;
    a[4]  = (_Float16)x1.x; a[5]  = (_Float16)x1.y; a[6]  = (_Float16)x1.z; a[7]  = (_Float16)x1.w;
    a[8]  = (_Float16)y0.x; a[9]  = (_Float16)y0.y; a[10] = (_Float16)y0.z; a[11] = (_Float16)y0.w;
    a[12] = (_Float16)y1.x; a[13] = (_Float16)y1.y; a[14] = (_Float16)y1.z; a[15] = (_Float16)y1.w;
    return a;
}

// softplus(beta=0.5, threshold=14): x if 0.5*x > 14 else 2*log1p(exp(0.5*x))
__device__ __forceinline__ float softplus_b05(float x) {
    float z = 0.5f * x;
    return (z > 14.0f) ? x : 2.0f * log1pf(__expf(z));
}

__global__ __launch_bounds__(256)
void veconv_fused_kernel(const float* __restrict__ node,
                         const float* __restrict__ rbf,
                         const float* __restrict__ edgef,
                         const int*   __restrict__ src,
                         const int*   __restrict__ dst,
                         const float* __restrict__ W1, const float* __restrict__ b1,
                         const float* __restrict__ W2, const float* __restrict__ b2,
                         const float* __restrict__ W3, const float* __restrict__ b3,
                         float* __restrict__ out,
                         int E, int numTiles)
{
    // Transposed f16 weights: B-fragment = 16 consecutive K at fixed n => one 32B LDS read.
    __shared__ __attribute__((aligned(32))) _Float16 sW1t[DIM][RBF_DIM]; // 16 KB
    __shared__ __attribute__((aligned(32))) _Float16 sW2t[DIM][DIM];     //  8 KB
    __shared__ __attribute__((aligned(32))) _Float16 sW3t[DIM][DIM];     //  8 KB
    __shared__ __attribute__((aligned(32))) _Float16 sH[8][16][DIM];     // 16 KB (per-wave h1 scratch)

    const int tid = threadIdx.x;

    // Cooperative weight load + transpose + f32->f16 (amortized over many tiles).
    for (int i = tid; i < RBF_DIM * DIM; i += 256) {
        int k = i / DIM, n = i - k * DIM;
        sW1t[n][k] = (_Float16)W1[i];
    }
    for (int i = tid; i < DIM * DIM; i += 256) {
        int k = i / DIM, n = i - k * DIM;
        sW2t[n][k] = (_Float16)W2[i];
        sW3t[n][k] = (_Float16)W3[i];
    }
    __syncthreads();

    const int wave = tid >> 5;
    const int lane = tid & 31;
    const int r16  = lane & 15;   // A-layout row / B,C,D-layout column (mod 16)
    const int half = lane >> 4;   // K sub-band select / D-layout M offset (+8)
    const int waveStride = (int)gridDim.x * 8;

    for (int tile = (int)blockIdx.x * 8 + wave; tile < numTiles; tile += waveStride) {
        int eA = tile * 16 + r16;                 // this lane's A-matrix edge row
        if (eA >= E) eA = E - 1;                  // clamp loads; stores guarded below
        const float* rbfRow = rbf   + (size_t)eA * RBF_DIM;
        const float* edgRow = edgef + (size_t)eA * DIM;

        // Hide HBM latency for the next tile this wave will touch.
        int nt = tile + waveStride;
        if (nt < numTiles) {
            int en = nt * 16 + r16; if (en >= E) en = E - 1;
            __builtin_prefetch(rbf + (size_t)en * RBF_DIM, 0, 1);
        }

        // ---------------- GEMM1: acc1 = rbf_tile[16x128] @ W1[128x64] ----------------
        v8f acc1[4] = {v8f{}, v8f{}, v8f{}, v8f{}};
        #pragma unroll
        for (int c = 0; c < 4; ++c) {             // K chunks of 32
            v16h A = load_A_f32(rbfRow, c, half);
            #pragma unroll
            for (int t = 0; t < 4; ++t) {         // N tiles of 16
                const _Float16* bp = &sW1t[16 * t + r16][32 * c + 16 * half];
                v16h B = *reinterpret_cast<const v16h*>(bp);
                acc1[t] = __builtin_amdgcn_wmma_f32_16x16x32_f16(
                    false, A, false, B, (short)0, acc1[t], false, false);
            }
        }

        // bias + softplus (fp32), store f16 row-major [edge_row][col] in wave scratch
        #pragma unroll
        for (int t = 0; t < 4; ++t) {
            int col = 16 * t + r16;
            float bc = b1[col];
            #pragma unroll
            for (int r = 0; r < 8; ++r) {
                int m = r + 8 * half;             // D layout: VGPR r -> row r (+8 for upper lanes)
                sH[wave][m][col] = (_Float16)softplus_b05(acc1[t][r] + bc);
            }
        }
        // same-wave DS ordering: compiler inserts s_wait_dscnt before dependent reads

        // ------- GEMM2: h = h1[16x64] @ W2[64x64]; GEMM3: ef' = edge_f @ W3 -------
        v8f acc2[4] = {v8f{}, v8f{}, v8f{}, v8f{}};
        v8f acc3[4] = {v8f{}, v8f{}, v8f{}, v8f{}};
        #pragma unroll
        for (int c = 0; c < 2; ++c) {             // K chunks of 32 (K=64)
            // A2 from LDS scratch (already f16, A layout = 16 contiguous K)
            const _Float16* hp = &sH[wave][r16][32 * c + 16 * half];
            v16h A2;
            #pragma unroll
            for (int i = 0; i < 8; ++i) { A2[i] = hp[i]; A2[8 + i] = hp[8 + i]; }
            v16h A3 = load_A_f32(edgRow, c, half);
            #pragma unroll
            for (int t = 0; t < 4; ++t) {
                int n = 16 * t + r16;
                v16h B2 = *reinterpret_cast<const v16h*>(&sW2t[n][32 * c + 16 * half]);
                v16h B3 = *reinterpret_cast<const v16h*>(&sW3t[n][32 * c + 16 * half]);
                acc2[t] = __builtin_amdgcn_wmma_f32_16x16x32_f16(
                    false, A2, false, B2, (short)0, acc2[t], false, false);
                acc3[t] = __builtin_amdgcn_wmma_f32_16x16x32_f16(
                    false, A3, false, B3, (short)0, acc3[t], false, false);
            }
        }

        // ---- gather node[src], combine, scatter-add to out[dst] (f32 atomics) ----
        int eBase = tile * 16 + 8 * half;
        int srcs[8], dsts[8];
        #pragma unroll
        for (int r = 0; r < 8; ++r) {
            int e = eBase + r;
            bool v = (e < E);
            srcs[r] = v ? src[e] : 0;
            dsts[r] = v ? dst[e] : -1;
        }
        #pragma unroll
        for (int t = 0; t < 4; ++t) {
            int col = 16 * t + r16;
            float b2c = b2[col], b3c = b3[col];
            #pragma unroll
            for (int r = 0; r < 8; ++r) {
                if (dsts[r] >= 0) {
                    float nv  = node[(size_t)srcs[r] * DIM + col];  // 12.8MB table: L2-resident
                    float val = nv * (acc2[t][r] + b2c) + (acc3[t][r] + b3c);
                    __hip_atomic_fetch_add(&out[(size_t)dsts[r] * DIM + col], val,
                                           __ATOMIC_RELAXED, __HIP_MEMORY_SCOPE_AGENT);
                }
            }
        }
    }
}

extern "C" void kernel_launch(void* const* d_in, const int* in_sizes, int n_in,
                              void* d_out, int out_size, void* d_ws, size_t ws_size,
                              hipStream_t stream) {
    const float* node  = (const float*)d_in[0];
    const float* rbf   = (const float*)d_in[1];
    const float* edgef = (const float*)d_in[2];
    const int*   src   = (const int*)  d_in[3];
    const int*   dst   = (const int*)  d_in[4];
    const float* W1    = (const float*)d_in[5];
    const float* b1    = (const float*)d_in[6];
    const float* W2    = (const float*)d_in[7];
    const float* b2    = (const float*)d_in[8];
    const float* W3    = (const float*)d_in[9];
    const float* b3    = (const float*)d_in[10];
    float* out = (float*)d_out;

    const int E = in_sizes[3];                 // src has E elements
    const int numTiles = (E + 15) / 16;

    // zero the scatter-destination (capture-legal memset node)
    hipMemsetAsync(out, 0, (size_t)out_size * sizeof(float), stream);

    int blocks = (numTiles + 7) / 8;           // 8 waves per block, 1 tile per wave per iter
    if (blocks > 1280) blocks = 1280;          // grid-stride; amortizes LDS weight staging
    veconv_fused_kernel<<<blocks, 256, 0, stream>>>(
        node, rbf, edgef, src, dst, W1, b1, W2, b2, W3, b3, out, E, numTiles);
}
